// fcn_8950711845741
// MI455X (gfx1250) — matrix operand order repaired and checked by reading, not compiled, not run on hardware
//
#include <hip/hip_runtime.h>
#include <hip/hip_bf16.h>

// ---------------------------------------------------------------------------
// NOSO SNN forward, restructured for MI455X (gfx1250, wave32, WMMA):
//   The dual-exponential traces are linear filters and commute with the
//   GEMMs, so both per-step GEMMs become one big parallel GEMM over all
//   300*256 = 76800 (t,b) rows, followed by cheap per-neuron scans for the
//   spike / one-spike (sav) gating.
//
// All WMMA operands are staged in *fragment-swizzled* layouts: each 16x32 A
// tile / 32x16 B tile is stored as [lane 0..31][16 halves], so a fragment
// load is one aligned contiguous 32B load per lane (1KB coalesced per wave),
// through explicit address_space(1) pointers -> global_load_b128, no flat.
// ---------------------------------------------------------------------------

typedef __attribute__((ext_vector_type(16))) _Float16 v16h;
typedef __attribute__((ext_vector_type(8)))  float    v8f;

#define AS1 __attribute__((address_space(1)))

#define T_STEPS 300
#define BATCH   256
#define N_IN    2312
#define KP      2336                 // N_IN padded to multiple of 32
#define KT1     (KP / 32)            // 73 K-tiles, layer 1
#define N_H     800
#define KT2     (N_H / 32)           // 25 K-tiles, layer 2
#define N_OUT   10
#define NOP     16                   // N_OUT padded to one WMMA N-tile
#define TB      (T_STEPS * BATCH)    // 76800 GEMM rows
#define DM      0.951229424500714f   // exp(-1/20)
#define DS      0.818730753077982f   // exp(-1/5)

// ---- global (AS1) memory helpers ------------------------------------------
__device__ __forceinline__ v16h gload_v16h(const _Float16* p) {
    return *(const v16h AS1*)(unsigned long long)p;
}
__device__ __forceinline__ float gload_f(const float* p) {
    return *(const float AS1*)(unsigned long long)p;
}
__device__ __forceinline__ void gstore_f(float* p, float v) {
    *(float AS1*)(unsigned long long)p = v;
}
__device__ __forceinline__ void gstore_h(_Float16* p, _Float16 v) {
    *(_Float16 AS1*)(unsigned long long)p = v;
}

// ---- fragment swizzle maps (CDNA5 wave32 WMMA f16, ISA 7.12.2) -------------
// A 16x32 tile: lane = m + 16*hs;  halves e=0..7 <-> k = hs*8+e,
//                                  halves e=8..15 <-> k = 16+hs*8+(e-8)
// inverse: given ki (0..31): hs=(ki>>3)&1, e=(ki&7)+((ki>>4)<<3)
__device__ __forceinline__ int a_swz(int m, int ki) {
    int hs = (ki >> 3) & 1;
    int e  = (ki & 7) + ((ki >> 4) << 3);
    return (m + 16 * hs) * 16 + e;           // offset in halves within 512-half tile
}
// B 32x16 tile: lane = n + 16*hs; same k order as A (e=0..7 <-> k=hs*8+e, e=8..15 <-> k=16+hs*8+(e-8))
__device__ __forceinline__ int b_swz(int n, int ki) {
    int hs = (ki >> 3) & 1;
    int e  = (ki & 7) + ((ki >> 4) << 3);
    return (n + 16 * hs) * 16 + e;
}

// ---------------------------------------------------------------------------
// K0: zero the 3 sum_sp accumulators at the tail of d_out
// ---------------------------------------------------------------------------
__global__ void k_zero_sums(float* __restrict__ p) {
    if (threadIdx.x < 3) gstore_f(p + threadIdx.x, 0.0f);
}

// ---------------------------------------------------------------------------
// K1: tiled transpose + f32->f16 of input [B, N_IN, T] -> swizzled A tiles
//     XtS tile index (rt, kt) with rt = r/16, r = t*BATCH + b; also reduces
//     sum(in_spike) into sumsp[0].
// ---------------------------------------------------------------------------
__global__ void __launch_bounds__(256)
k_transpose_input(const float* __restrict__ in, _Float16* __restrict__ XtS,
                  float* __restrict__ sumsp) {
    __shared__ float tile[32][33];
    __shared__ float red[256];
    const int b   = blockIdx.z;
    const int i0  = blockIdx.x * 32;       // one K-tile (kt = blockIdx.x)
    const int t0  = blockIdx.y * 32;
    const int tid = threadIdx.x;
    float local = 0.0f;
#pragma unroll
    for (int l = 0; l < 4; ++l) {
        int lin = tid + l * 256;
        int ii = lin >> 5, tt = lin & 31;  // consecutive tid -> consecutive t (coalesced reads)
        float v = 0.0f;
        int gi = i0 + ii, gt = t0 + tt;
        if (gi < N_IN && gt < T_STEPS)
            v = gload_f(&in[((size_t)b * N_IN + gi) * T_STEPS + gt]);
        tile[ii][tt] = v;
        local += v;
    }
    __syncthreads();
    const int kt = blockIdx.x;
    const int m  = b & 15;
    const int bt = b >> 4;
#pragma unroll
    for (int l = 0; l < 4; ++l) {
        int lin = tid + l * 256;
        int tt = lin >> 5, ki = lin & 31;
        int gt = t0 + tt;
        if (gt < T_STEPS) {
            size_t rt  = (size_t)gt * 16 + bt;               // r/16
            size_t off = (rt * KT1 + kt) * 512 + a_swz(m, ki);
            gstore_h(XtS + off, (_Float16)tile[ki][tt]);
        }
    }
    red[tid] = local;
    __syncthreads();
    for (int s = 128; s > 0; s >>= 1) {
        if (tid < s) red[tid] += red[tid + s];
        __syncthreads();
    }
    if (tid == 0) atomicAdd(sumsp, red[0]);
}

// ---------------------------------------------------------------------------
// K2: w1 [N_H][N_IN] f32 -> swizzled B tiles W1S [nt=h/16][kt][lane][16]
// ---------------------------------------------------------------------------
__global__ void k_conv_w1(const float* __restrict__ w1, _Float16* __restrict__ W1S) {
    int idx = blockIdx.x * 256 + threadIdx.x;
    if (idx >= N_H * KP) return;
    int h = idx / KP, i = idx % KP;
    float v = (i < N_IN) ? gload_f(&w1[(size_t)h * N_IN + i]) : 0.0f;
    size_t off = (((size_t)(h >> 4) * KT1 + (i >> 5)) * 512) + b_swz(h & 15, i & 31);
    gstore_h(W1S + off, (_Float16)v);
}

// K3: w2 [10][N_H] -> swizzled B tiles W2S [1 n-tile][kt][lane][16], rows 10..15 zero
__global__ void k_conv_w2(const float* __restrict__ w2, _Float16* __restrict__ W2S) {
    int idx = blockIdx.x * 256 + threadIdx.x;
    if (idx >= NOP * N_H) return;
    int o = idx / N_H, h = idx % N_H;
    float v = (o < N_OUT) ? gload_f(&w2[(size_t)o * N_H + h]) : 0.0f;
    size_t off = ((size_t)(h >> 5) * 512) + b_swz(o, h & 31);
    gstore_h(W2S + off, (_Float16)v);
}

// ---------------------------------------------------------------------------
// K4: GEMM1  Y[TB, N_H] = X * W1^T   (f16 fragments, f32 accumulate)
//     wave tile: 2 M-tiles x 5 N-tiles (10 wmma / K-step)
//     workgroup: 8 waves = 4(M) x 2(N) -> block tile 128 x 160
// ---------------------------------------------------------------------------
__global__ void __launch_bounds__(256)
k_gemm1(const _Float16* __restrict__ XtS, const _Float16* __restrict__ W1S,
        float* __restrict__ Y) {
    const int tid  = threadIdx.x;
    const int lane = tid & 31;
    const int wave = tid >> 5;
    const int wm   = wave & 3;
    const int wn   = wave >> 2;

    const int mt0 = blockIdx.x * 8 + wm * 2;     // first of 2 M-tiles
    const int nt0 = blockIdx.y * 10 + wn * 5;    // first of 5 N-tiles

    const _Float16* aptr0 = XtS + ((size_t)mt0 * KT1) * 512 + lane * 16;
    const _Float16* aptr1 = aptr0 + (size_t)KT1 * 512;
    const _Float16* bptr[5];
#pragma unroll
    for (int j = 0; j < 5; ++j)
        bptr[j] = W1S + ((size_t)(nt0 + j) * KT1) * 512 + lane * 16;

    v8f zero = {};
    v8f acc[2][5];
#pragma unroll
    for (int i = 0; i < 2; ++i)
#pragma unroll
        for (int j = 0; j < 5; ++j) acc[i][j] = zero;

    for (int kt = 0; kt < KT1; ++kt) {
        __builtin_prefetch(aptr0 + 1024, 0, 1);   // global_prefetch: next A tiles
        v16h a0 = gload_v16h(aptr0);
        v16h a1 = gload_v16h(aptr1);
#pragma unroll
        for (int j = 0; j < 5; ++j) {
            v16h b = gload_v16h(bptr[j]);
            acc[0][j] = __builtin_amdgcn_wmma_f32_16x16x32_f16(
                false, a0, false, b, (short)0, acc[0][j], false, false);
            acc[1][j] = __builtin_amdgcn_wmma_f32_16x16x32_f16(
                false, a1, false, b, (short)0, acc[1][j], false, false);
            bptr[j] += 512;
        }
        aptr0 += 512;
        aptr1 += 512;
    }

    // C/D layout: VGPR p, lanes 0-15 -> row p, lanes 16-31 -> row p+8; col = lane&15
    const int m      = lane & 15;
    const int rowOff = (lane >> 4) * 8;
#pragma unroll
    for (int i = 0; i < 2; ++i)
#pragma unroll
        for (int j = 0; j < 5; ++j)
#pragma unroll
            for (int p = 0; p < 8; ++p) {
                int row = (mt0 + i) * 16 + rowOff + p;
                int col = (nt0 + j) * 16 + m;
                gstore_f(&Y[(size_t)row * N_H + col], acc[i][j][p]);
            }
}

// ---------------------------------------------------------------------------
// K5: layer-1 temporal scan. One thread per (b,h). Filters Y (linear part),
//     threshold + one-spike sav gate, writes spk1 {0,1} f16 in swizzled
//     A-tile layout for GEMM2.
// ---------------------------------------------------------------------------
__global__ void __launch_bounds__(256)
k_scan1(const float* __restrict__ Y, _Float16* __restrict__ S1S,
        float* __restrict__ sumsp) {
    __shared__ float red[256];
    const int idx = blockIdx.x * 256 + threadIdx.x;  // exactly BATCH*N_H threads
    const int b = idx / N_H, h = idx % N_H;
    const float* yp = Y + (size_t)b * N_H + h;
    // swizzled spk1 address: rt = t*16 + (b>>4), kt = h>>5
    _Float16* sp = S1S + (((size_t)(b >> 4) * KT2 + (h >> 5)) * 512) + a_swz(b & 15, h & 31);
    const size_t ystride = (size_t)BATCH * N_H;       // per-t stride in Y
    const size_t sstride = (size_t)16 * KT2 * 512;    // per-t stride in S1S (= BATCH*N_H)
    float vm = 0.f, vs = 0.f, sav = 1.f, cnt = 0.f;
    for (int t = 0; t < T_STEPS; ++t) {
        float y = gload_f(yp + (size_t)t * ystride);
        vm = DM * vm + y;
        vs = DS * vs + y;
        float u = sav * (vm - vs);
        float s = (u > 1.0f) ? 1.0f : 0.0f;   // act_fun(u - THRESH)
        sav *= (1.0f - s);
        gstore_h(sp + (size_t)t * sstride, (_Float16)s);
        cnt += s;
    }
    red[threadIdx.x] = cnt;
    __syncthreads();
    for (int s2 = 128; s2 > 0; s2 >>= 1) {
        if (threadIdx.x < s2) red[threadIdx.x] += red[threadIdx.x + s2];
        __syncthreads();
    }
    if (threadIdx.x == 0) atomicAdd(sumsp + 1, red[0]);
}

// ---------------------------------------------------------------------------
// K6: GEMM2  Z[TB, 16] = spk1 * W2^T   (N padded 10->16)
//     wave tile: 4 M-tiles x 1 N-tile; workgroup 8 waves -> 512 rows/block
// ---------------------------------------------------------------------------
__global__ void __launch_bounds__(256)
k_gemm2(const _Float16* __restrict__ S1S, const _Float16* __restrict__ W2S,
        float* __restrict__ Z) {
    const int tid  = threadIdx.x;
    const int lane = tid & 31;
    const int wave = tid >> 5;
    const int mt0  = (blockIdx.x * 8 + wave) * 4;

    const _Float16* aptr[4];
#pragma unroll
    for (int i = 0; i < 4; ++i)
        aptr[i] = S1S + ((size_t)(mt0 + i) * KT2) * 512 + lane * 16;
    const _Float16* bptr = W2S + lane * 16;

    v8f zero = {};
    v8f acc[4];
#pragma unroll
    for (int i = 0; i < 4; ++i) acc[i] = zero;

    for (int kt = 0; kt < KT2; ++kt) {
        v16h b = gload_v16h(bptr);
#pragma unroll
        for (int i = 0; i < 4; ++i) {
            v16h a = gload_v16h(aptr[i]);
            acc[i] = __builtin_amdgcn_wmma_f32_16x16x32_f16(
                false, a, false, b, (short)0, acc[i], false, false);
            aptr[i] += 512;
        }
        bptr += 512;
    }
    const int m      = lane & 15;
    const int rowOff = (lane >> 4) * 8;
#pragma unroll
    for (int i = 0; i < 4; ++i)
#pragma unroll
        for (int p = 0; p < 8; ++p)
            gstore_f(&Z[(size_t)((mt0 + i) * 16 + rowOff + p) * NOP + m], acc[i][p]);
}

// ---------------------------------------------------------------------------
// K7: layer-2 scan. One thread per (b,o). Produces out_t, out_u, spk2 count.
// ---------------------------------------------------------------------------
__global__ void __launch_bounds__(256)
k_scan2(const float* __restrict__ Z, float* __restrict__ out,
        float* __restrict__ sumsp) {
    __shared__ float red[256];
    const int idx = blockIdx.x * 256 + threadIdx.x;
    float cnt = 0.f;
    if (idx < BATCH * N_OUT) {
        const int b = idx / N_OUT, o = idx % N_OUT;
        const size_t stride = (size_t)BATCH * NOP;
        const float* zp = Z + (size_t)b * NOP + o;
        float vm = 0.f, vs = 0.f, sav = 1.f;
        float ot = (float)T_STEPS, ou = 0.f;
        for (int t = 0; t < T_STEPS; ++t) {
            float z = gload_f(zp + (size_t)t * stride);
            vm = DM * vm + z;
            vs = DS * vs + z;
            float u = sav * (vm - vs);
            float s = (u > 1.0f) ? 1.0f : 0.0f;
            sav *= (1.0f - s);
            ot += s * ((float)t - (float)T_STEPS);  // dodt(spk2, step - NUM_STEPS)
            ou += s * u;
            cnt += s;
        }
        gstore_f(&out[idx], ot);                        // out_t
        gstore_f(&out[BATCH * N_OUT + idx], ou);        // out_u
    }
    red[threadIdx.x] = cnt;
    __syncthreads();
    for (int s2 = 128; s2 > 0; s2 >>= 1) {
        if (threadIdx.x < s2) red[threadIdx.x] += red[threadIdx.x + s2];
        __syncthreads();
    }
    if (threadIdx.x == 0) atomicAdd(sumsp + 2, red[0]);
}

// ---------------------------------------------------------------------------
// Workspace layout (bytes, 256-aligned, liveness-based aliasing):
//   W1S  @ 0           : 50*73*1024      =   3,737,600
//   W2S  @ 3,737,600   : 25*1024        =       25,600
//   XtS  @ 3,763,200   : 4800*73*1024   =  358,809,600  (dead after GEMM1)
//   S1S  @ 3,763,200   : 4800*25*1024   =  122,880,000  (aliases XtS)
//   Z    @ 126,643,200 : 76800*16*4     =    4,915,200  (aliases XtS tail)
//   Y    @ 362,572,800 : 76800*800*4    =  245,760,000
// ---------------------------------------------------------------------------
extern "C" void kernel_launch(void* const* d_in, const int* in_sizes, int n_in,
                              void* d_out, int out_size, void* d_ws, size_t ws_size,
                              hipStream_t stream) {
    const float* d_input = (const float*)d_in[0];  // [B, 2, 34, 34, T] = [B, N_IN, T]
    const float* d_w1    = (const float*)d_in[1];  // [N_H, N_IN]
    const float* d_w2    = (const float*)d_in[2];  // [N_OUT, N_H]
    float*       out     = (float*)d_out;          // out_t(2560) | out_u(2560) | sum_sp(3)

    char* ws = (char*)d_ws;
    _Float16* W1S = (_Float16*)(ws);
    _Float16* W2S = (_Float16*)(ws + 3737600);
    _Float16* XtS = (_Float16*)(ws + 3763200);
    _Float16* S1S = (_Float16*)(ws + 3763200);      // aliases XtS (dead by then)
    float*    Z   = (float*)   (ws + 126643200);    // aliases XtS tail
    float*    Y   = (float*)   (ws + 362572800);
    float*    sums = out + 2 * BATCH * N_OUT;       // d_out[5120..5122]

    k_zero_sums<<<1, 32, 0, stream>>>(sums);
    k_conv_w1<<<(N_H * KP + 255) / 256, 256, 0, stream>>>(d_w1, W1S);
    k_conv_w2<<<(NOP * N_H + 255) / 256, 256, 0, stream>>>(d_w2, W2S);
    k_transpose_input<<<dim3(KT1, (T_STEPS + 31) / 32, BATCH), 256, 0, stream>>>(
        d_input, XtS, sums);
    k_gemm1<<<dim3(TB / 128, N_H / 160), 256, 0, stream>>>(XtS, W1S, Y);
    k_scan1<<<(BATCH * N_H) / 256, 256, 0, stream>>>(Y, S1S, sums);
    k_gemm2<<<TB / 512, 256, 0, stream>>>(S1S, W2S, Z);
    k_scan2<<<(BATCH * N_OUT + 255) / 256, 256, 0, stream>>>(Z, out, sums);
}
